// GramSpecMoEBlock_44693429682386
// MI455X (gfx1250) — compile-verified, matrix operand order checked by reading, not run on hardware
//
#include <hip/hip_runtime.h>

// ---------------------------------------------------------------------------
// GramSpecMoE block for MI455X (gfx1250, wave32, WMMA).
// - Heavy GEMMs: V_WMMA_F32_16X16X32_BF16, f32 accumulate, convert-on-load.
// - MoE intermediate activations stored as bf16 -> down-proj A tiles staged
//   with GLOBAL_LOAD_ASYNC_TO_LDS_B128 (ASYNCcnt-tracked).
// - Exact top-2 sparsity (zero-weight experts contribute exactly 0).
// ---------------------------------------------------------------------------

#define BDIM   2
#define SEQ    2048
#define TOK    4096      // B*S
#define HID    1024
#define GRU    1024
#define FFN    2048
#define NEXP   8

typedef __attribute__((ext_vector_type(16))) __bf16 v16bf;
typedef __attribute__((ext_vector_type(2)))  __bf16 v2bf;
typedef __attribute__((ext_vector_type(8)))  float  v8f;
typedef int v4i_gcc __attribute__((vector_size(16)));   // matches builtin's V4i

union FragU { v16bf v; unsigned u[8]; };

// Packed f32x2 -> bf16x2 via native __bf16 conversion (compiler picks the
// target's cvt instruction; RNE rounding).
__device__ __forceinline__ unsigned pk_bf16(float lo, float hi) {
    union { v2bf v; unsigned u; } cv;
    cv.v = v2bf{(__bf16)lo, (__bf16)hi};
    return cv.u;
}

// Async global->LDS b128 copy (CDNA5 GLOBAL_LOAD_ASYNC_TO_LDS_B128).
// Signature (from hipcc diagnostic): (v4i AS1* src, v4i AS3* dst, imm, imm).
#if __has_builtin(__builtin_amdgcn_global_load_async_to_lds_b128)
#define HAVE_ASYNC_LDS 1
__device__ __forceinline__ void async_cp16(const void* g, void* l) {
    __builtin_amdgcn_global_load_async_to_lds_b128(
        (__attribute__((address_space(1))) v4i_gcc*)(g),
        (__attribute__((address_space(3))) v4i_gcc*)(l), 0, 0);
}
#else
#define HAVE_ASYNC_LDS 0
#endif

__device__ __forceinline__ void wait_async0() {
#if __has_builtin(__builtin_amdgcn_s_wait_asynccnt)
    __builtin_amdgcn_s_wait_asynccnt(0);
#else
    asm volatile("s_wait_asynccnt 0" ::: "memory");
#endif
}

__device__ __forceinline__ v8f wmma_bf16(v16bf a, v16bf b, v8f c) {
    return __builtin_amdgcn_wmma_f32_16x16x32_bf16(
        false, a, false, b, (short)0, c, false, false);
}

#define BM 64
#define BN 64
#define BK 32

// Load one 16x32 bf16 fragment from an LDS tile laid out [rows][BK] ushort.
// ISA 7.12.2 16-bit A layout (B^T uses the identical mapping).
__device__ __forceinline__ v16bf load_frag(const unsigned short* tile, int row0, int lane) {
    int half = lane >> 4;
    const unsigned* rp = (const unsigned*)(tile + (size_t)(row0 + (lane & 15)) * BK);
    FragU f;
#pragma unroll
    for (int j = 0; j < 8; ++j) {
        int uidx = (j & 3) + 4 * half + 8 * (j >> 2);
        f.u[j] = rp[uidx];
    }
    return f.v;
}

// ---------------------------------------------------------------------------
// Generic tiled GEMM:  C(M,N) = A(M,K) * W(N,K)^T   (weights row-major [N,K])
// MODE 0: A f32 dense, C f32 plain store           (xw / expr projections)
// MODE 1: A f32 gathered rows, dual-W, silu(g)*u -> C bf16 (token-local interm)
// MODE 2: A bf16 token-local interm (async LDS staging), scatter-add
//         out[idx[m]] += wt[m]*acc  (C f32)
// ---------------------------------------------------------------------------
template <int MODE>
__global__ void __launch_bounds__(256)
gemm_bf16(const void* __restrict__ Avoid, int lda,
          const float* __restrict__ W0, const float* __restrict__ W1, int K,
          void* __restrict__ Cvoid, int ldc,
          const int* __restrict__ countPtr,
          const int* __restrict__ idx,
          const float* __restrict__ wt) {
    __shared__ unsigned short As[BM][BK];
    __shared__ unsigned short Bs0[BN][BK];
    __shared__ unsigned short Bs1[(MODE == 1) ? BN : 1][BK];

    const int bn = blockIdx.x, bm = blockIdx.y;
    int count = 0;
    if (MODE != 0) {
        count = *countPtr;
        if (bm * BM >= count) return;
    }

    const int tid  = threadIdx.x;
    const int lrow = tid >> 2;          // 0..63
    const int lk   = (tid & 3) * 8;     // 0,8,16,24

    int am = bm * BM + lrow;
    int arow;
    if (MODE == 1) arow = (am < count) ? (idx ? idx[am] : am) : 0;
    else           arow = am;

    const float*          Apf = (MODE == 2) ? nullptr
                              : (const float*)Avoid + (size_t)arow * lda + lk;
    const unsigned short* Aph = (MODE == 2)
                              ? (const unsigned short*)Avoid + (size_t)arow * lda + lk
                              : nullptr;
    const float* W0p = W0 + (size_t)(bn * BN + lrow) * K + lk;
    const float* W1p = (MODE == 1) ? (W1 + (size_t)(bn * BN + lrow) * K + lk) : nullptr;

    const int wave = tid >> 5, lane = tid & 31;
    const int wm = wave & 3;            // M sub-tile (16 rows)
    const int wn = wave >> 2;           // N sub-block (32 cols = 2 wmma tiles)

    v8f acc0 = {0.f,0.f,0.f,0.f,0.f,0.f,0.f,0.f};
    v8f acc1 = {0.f,0.f,0.f,0.f,0.f,0.f,0.f,0.f};
    v8f acu0 = {0.f,0.f,0.f,0.f,0.f,0.f,0.f,0.f};
    v8f acu1 = {0.f,0.f,0.f,0.f,0.f,0.f,0.f,0.f};

    for (int kb = 0; kb < K; kb += BK) {
        // ---------------- stage A tile ----------------
        if constexpr (MODE == 2) {
            // A already bf16: pure copy -> async DMA straight into LDS.
#if HAVE_ASYNC_LDS
            async_cp16(Aph + kb, &As[lrow][lk]);
#else
            *(uint4*)&As[lrow][lk] = *(const uint4*)(Aph + kb);
#endif
        } else {
            float4 a0 = *(const float4*)(Apf + kb);
            float4 a1 = *(const float4*)(Apf + kb + 4);
            unsigned* ar = (unsigned*)&As[lrow][lk];
            ar[0] = pk_bf16(a0.x, a0.y); ar[1] = pk_bf16(a0.z, a0.w);
            ar[2] = pk_bf16(a1.x, a1.y); ar[3] = pk_bf16(a1.z, a1.w);
        }
        // ---------------- stage W tiles (f32 -> bf16 in flight) -------------
        {
            float4 b0 = *(const float4*)(W0p + kb);
            float4 b1 = *(const float4*)(W0p + kb + 4);
            unsigned* br = (unsigned*)&Bs0[lrow][lk];
            br[0] = pk_bf16(b0.x, b0.y); br[1] = pk_bf16(b0.z, b0.w);
            br[2] = pk_bf16(b1.x, b1.y); br[3] = pk_bf16(b1.z, b1.w);
        }
        if constexpr (MODE == 1) {
            float4 c0 = *(const float4*)(W1p + kb);
            float4 c1 = *(const float4*)(W1p + kb + 4);
            unsigned* cr = (unsigned*)&Bs1[lrow][lk];
            cr[0] = pk_bf16(c0.x, c0.y); cr[1] = pk_bf16(c0.z, c0.w);
            cr[2] = pk_bf16(c1.x, c1.y); cr[3] = pk_bf16(c1.z, c1.w);
        }
        if (kb + BK < K) {
            __builtin_prefetch(W0p + kb + BK, 0, 0);   // global_prefetch_b8
        }
#if HAVE_ASYNC_LDS
        if constexpr (MODE == 2) wait_async0();        // s_wait_asynccnt 0
#endif
        __syncthreads();

        // ---------------- WMMA ----------------
        v16bf fa  = load_frag(&As[0][0],  wm * 16,      lane);
        v16bf fb0 = load_frag(&Bs0[0][0], wn * 32,      lane);
        v16bf fb1 = load_frag(&Bs0[0][0], wn * 32 + 16, lane);
        acc0 = wmma_bf16(fa, fb0, acc0);
        acc1 = wmma_bf16(fa, fb1, acc1);
        if constexpr (MODE == 1) {
            v16bf fc0 = load_frag(&Bs1[0][0], wn * 32,      lane);
            v16bf fc1 = load_frag(&Bs1[0][0], wn * 32 + 16, lane);
            acu0 = wmma_bf16(fa, fc0, acu0);
            acu1 = wmma_bf16(fa, fc1, acu1);
        }
        __syncthreads();
    }

    // ---- store (C layout: VGPR r -> M = r + 8*(lane>=16), N = lane&15) -----
    const int half = lane >> 4, nc = lane & 15;
    const int n0 = bn * BN + wn * 32;
    if constexpr (MODE == 0) {
        float* C = (float*)Cvoid;
#pragma unroll
        for (int r = 0; r < 8; ++r) {
            int m = bm * BM + wm * 16 + r + 8 * half;
            C[(size_t)m * ldc + n0 + nc]      = acc0[r];
            C[(size_t)m * ldc + n0 + 16 + nc] = acc1[r];
        }
    } else if constexpr (MODE == 1) {
        __bf16* C = (__bf16*)Cvoid;                   // bf16 interm
#pragma unroll
        for (int r = 0; r < 8; ++r) {
            int m = bm * BM + wm * 16 + r + 8 * half;
            if (m < count) {
                float g0 = acc0[r], g1 = acc1[r];
                float o0 = (g0 / (1.f + __expf(-g0))) * acu0[r];   // silu(g)*u
                float o1 = (g1 / (1.f + __expf(-g1))) * acu1[r];
                C[(size_t)m * ldc + n0 + nc]      = (__bf16)o0;
                C[(size_t)m * ldc + n0 + 16 + nc] = (__bf16)o1;
            }
        }
    } else {
        float* C = (float*)Cvoid;
#pragma unroll
        for (int r = 0; r < 8; ++r) {
            int m = bm * BM + wm * 16 + r + 8 * half;
            if (m < count) {
                int   orow = idx ? idx[m] : m;
                float sc   = wt ? wt[m] : 1.f;
                float* o = C + (size_t)orow * ldc + n0 + nc;
                o[0]  += sc * acc0[r];
                o[16] += sc * acc1[r];
            }
        }
    }
}

// ---------------------------------------------------------------------------
// Small kernels
// ---------------------------------------------------------------------------
__global__ void k_init(int* counts) {
    int i = threadIdx.x;
    if (i < NEXP) counts[i] = 0;
    if (i == NEXP) counts[NEXP] = TOK;   // shared-expert "count"
}

__global__ void k_zero(float* p, int n) {
    int i = blockIdx.x * 256 + threadIdx.x;
    if (i < n) p[i] = 0.f;
}

__global__ void k_mean(const float* __restrict__ hid, float* __restrict__ mean) {
    int id = blockIdx.x * 256 + threadIdx.x;       // 0..2047
    int b = id >> 10, h = id & 1023;
    const float* p = hid + ((size_t)b * SEQ) * HID + h;
    float s = 0.f;
    for (int ss = 0; ss < SEQ; ++ss) s += p[(size_t)ss * HID];
    mean[id] = s * (1.f / (float)SEQ);
}

__global__ void k_h0(const float* __restrict__ mean, const float* __restrict__ ctx_w,
                     const float* __restrict__ ctx_b, float* __restrict__ h0) {
    int id = blockIdx.x * 256 + threadIdx.x;       // 0..2047
    int b = id >> 10, g = id & 1023;
    const float* m = mean + b * HID;
    const float* w = ctx_w + (size_t)g * HID;
    float s = ctx_b[g];
    for (int k = 0; k < HID; ++k) s += m[k] * w[k];
    h0[id] = s;
}

// Sequential GRU scan: one workgroup per batch (the two scans are independent).
__global__ void __launch_bounds__(1024)
k_gru(const float* __restrict__ xw, const float* __restrict__ whh,
      const float* __restrict__ h0, float* __restrict__ routing) {
    __shared__ float h[GRU];
    const int b = blockIdx.x, g = threadIdx.x;
    h[g] = h0[b * GRU + g];
    __syncthreads();
    const float* wr = whh + (size_t)g * GRU;
    const float* wz = whh + (size_t)(GRU + g) * GRU;
    const float* wn = whh + (size_t)(2 * GRU + g) * GRU;
    for (int s = 0; s < SEQ; ++s) {
        size_t t = (size_t)(b * SEQ + s);
        const float* xr = xw + t * (3 * GRU);
        float hr = 0.f, hz = 0.f, hn = 0.f;
        for (int k = 0; k < GRU; ++k) {
            float hv = h[k];
            hr += wr[k] * hv; hz += wz[k] * hv; hn += wn[k] * hv;
        }
        float rg = 1.f / (1.f + __expf(-(xr[g] + hr)));
        float zg = 1.f / (1.f + __expf(-(xr[GRU + g] + hz)));
        float ng = tanhf(xr[2 * GRU + g] + rg * hn);
        float hnew = (1.f - zg) * ng + zg * h[g];
        __syncthreads();
        h[g] = hnew;
        routing[t * GRU + g] = hnew;
        __syncthreads();
    }
}

// Per-token: l2-normalize routing, gram penalty partial, cosine scores.
__global__ void k_token_stats(const float* __restrict__ routing, const float* __restrict__ expr,
                              float* __restrict__ cd, float* __restrict__ pen) {
    __shared__ float r[1024];
    __shared__ float x[1024];
    __shared__ float red[NEXP * 3];
    __shared__ float gpart[64][4];
    __shared__ float diff[NEXP][NEXP];
    __shared__ float rowpen[NEXP];
    const int t = blockIdx.x, tid = threadIdx.x;
    const float* rp = routing + (size_t)t * 1024;
    const float* xp = expr + (size_t)t * 1024;
    for (int i = tid; i < 1024; i += 256) { r[i] = rp[i]; x[i] = xp[i]; }
    __syncthreads();

    const int wave = tid >> 5, lane = tid & 31;    // wave == expert
    float srr = 0.f, sxx = 0.f, sxr = 0.f;
    for (int i = lane; i < 128; i += 32) {
        float rv = r[wave * 128 + i], xv = x[wave * 128 + i];
        srr += rv * rv; sxx += xv * xv; sxr += xv * rv;
    }
    for (int off = 16; off; off >>= 1) {
        srr += __shfl_down(srr, off);
        sxx += __shfl_down(sxx, off);
        sxr += __shfl_down(sxr, off);
    }
    if (lane == 0) { red[wave*3] = srr; red[wave*3+1] = sxx; red[wave*3+2] = sxr; }
    __syncthreads();
    srr = red[wave*3]; sxx = red[wave*3+1]; sxr = red[wave*3+2];
    float nr  = sqrtf(srr);
    float inv = 1.f / fmaxf(nr, 1e-12f);
    for (int i = lane; i < 128; i += 32) r[wave * 128 + i] *= inv;  // normalize
    if (lane == 0) {
        float num = sxr * inv;
        float den = fmaxf(sqrtf(sxx), 1e-8f) * fmaxf(nr * inv, 1e-8f);
        cd[(size_t)t * NEXP + wave] = 1.f - num / den;
    }
    __syncthreads();

    int p = tid >> 2, sub = tid & 3;
    int e = p >> 3, f = p & 7;
    float acc = 0.f;
    for (int i = sub * 32; i < sub * 32 + 32; ++i) acc += r[e * 128 + i] * r[f * 128 + i];
    gpart[p][sub] = acc;
    __syncthreads();
    if (tid < 64) {
        float d = gpart[tid][0] + gpart[tid][1] + gpart[tid][2] + gpart[tid][3];
        if ((tid >> 3) == (tid & 7)) d -= 1.f;
        diff[tid >> 3][tid & 7] = d;
    }
    __syncthreads();
    if (tid < NEXP) {
        float ss = 0.f;
        for (int j = 0; j < NEXP; ++j) ss += diff[tid][j] * diff[tid][j];
        float ri = 1.f / fmaxf(sqrtf(ss), 1e-12f);
        rowpen[tid] = ss * ri * ri;
    }
    __syncthreads();
    if (tid == 0) {
        float s = 0.f;
        for (int j = 0; j < NEXP; ++j) s += rowpen[j];
        pen[t] = s;
    }
}

__global__ void k_pen_reduce(const float* __restrict__ pen, float* __restrict__ penalty) {
    __shared__ float s[256];
    float a = 0.f;
    for (int i = threadIdx.x; i < TOK; i += 256) a += pen[i];
    s[threadIdx.x] = a;
    __syncthreads();
    for (int off = 128; off; off >>= 1) {
        if (threadIdx.x < off) s[threadIdx.x] += s[threadIdx.x + off];
        __syncthreads();
    }
    if (threadIdx.x == 0) penalty[0] = s[0] * (1.f / (float)TOK);
}

__global__ void k_topk(const float* __restrict__ cd, const float* __restrict__ penalty,
                       int* __restrict__ counts, int* __restrict__ tok, float* __restrict__ wt) {
    int t = blockIdx.x * 256 + threadIdx.x;
    if (t >= TOK) return;
    float pf = 1.f + penalty[0];
    float s[NEXP];
#pragma unroll
    for (int e = 0; e < NEXP; ++e) s[e] = cd[(size_t)t * NEXP + e] * pf;
    int e0 = 0; float v0 = s[0];
#pragma unroll
    for (int e = 1; e < NEXP; ++e) if (s[e] > v0) { v0 = s[e]; e0 = e; }
    int e1 = -1; float v1 = -3.4e38f;
#pragma unroll
    for (int e = 0; e < NEXP; ++e) if (e != e0 && s[e] > v1) { v1 = s[e]; e1 = e; }
    float w0 = 1.f / (1.f + __expf(v1 - v0));
    float w1 = 1.f - w0;
    int s0 = atomicAdd(&counts[e0], 1); tok[e0 * TOK + s0] = t; wt[e0 * TOK + s0] = w0;
    int s1 = atomicAdd(&counts[e1], 1); tok[e1 * TOK + s1] = t; wt[e1 * TOK + s1] = w1;
}

// ---------------------------------------------------------------------------
extern "C" void kernel_launch(void* const* d_in, const int* in_sizes, int n_in,
                              void* d_out, int out_size, void* d_ws, size_t ws_size,
                              hipStream_t stream) {
    const float* hid    = (const float*)d_in[0];
    const float* w_ih   = (const float*)d_in[1];
    const float* w_hh   = (const float*)d_in[2];
    const float* expr_w = (const float*)d_in[3];
    const float* ctx_w  = (const float*)d_in[4];
    const float* ctx_b  = (const float*)d_in[5];
    const float* gate_w = (const float*)d_in[6];
    const float* up_w   = (const float*)d_in[7];
    const float* down_w = (const float*)d_in[8];
    const float* sgw    = (const float*)d_in[9];
    const float* suw    = (const float*)d_in[10];
    const float* sdw    = (const float*)d_in[11];
    float* out = (float*)d_out;

    // ---- workspace carve-up (~101 MB) ----
    float* ws = (float*)d_ws;
    size_t off = 0;
    float* xw      = ws + off; off += (size_t)TOK * 3 * GRU;       // 50.3 MB
    float* routing = ws + off; off += (size_t)TOK * GRU;           // 16.8 MB
    float* expr    = ws + off; off += (size_t)TOK * GRU;           // 16.8 MB
    unsigned short* interm = (unsigned short*)(ws + off);
    off += (size_t)TOK * FFN / 2;                                  // 16.8 MB (bf16)
    float* meanB   = ws + off; off += BDIM * HID;
    float* h0      = ws + off; off += BDIM * GRU;
    float* cd      = ws + off; off += (size_t)TOK * NEXP;
    float* pen     = ws + off; off += TOK;
    float* penalty = ws + off; off += 16;
    float* wtbuf   = ws + off; off += (size_t)NEXP * TOK;
    int*   counts  = (int*)(ws + off); off += 16;
    int*   tok     = (int*)(ws + off); off += (size_t)NEXP * TOK;
    (void)in_sizes; (void)n_in; (void)out_size; (void)ws_size;

    k_init<<<1, 32, 0, stream>>>(counts);
    k_zero<<<(TOK * HID + 255) / 256, 256, 0, stream>>>(out, TOK * HID);
    k_mean<<<8, 256, 0, stream>>>(hid, meanB);
    k_h0<<<8, 256, 0, stream>>>(meanB, ctx_w, ctx_b, h0);

    // xw = hid @ w_ih^T  [4096 x 3072]
    gemm_bf16<0><<<dim3((3 * GRU) / BN, TOK / BM), 256, 0, stream>>>(
        hid, HID, w_ih, nullptr, HID, xw, 3 * GRU, nullptr, nullptr, nullptr);

    // sequential GRU scan (independent per batch)
    k_gru<<<BDIM, 1024, 0, stream>>>(xw, w_hh, h0, routing);

    // expr = hid @ expr_w^T  [4096 x 1024]
    gemm_bf16<0><<<dim3(GRU / BN, TOK / BM), 256, 0, stream>>>(
        hid, HID, expr_w, nullptr, HID, expr, GRU, nullptr, nullptr, nullptr);

    k_token_stats<<<TOK, 256, 0, stream>>>(routing, expr, cd, pen);
    k_pen_reduce<<<1, 256, 0, stream>>>(pen, penalty);
    k_topk<<<TOK / 256, 256, 0, stream>>>(cd, penalty, counts, tok, wtbuf);

    // shared expert (dense, weight 1, counts[NEXP] == TOK)
    gemm_bf16<1><<<dim3(FFN / BN, TOK / BM), 256, 0, stream>>>(
        hid, HID, sgw, suw, HID, interm, FFN, counts + NEXP, nullptr, nullptr);
    gemm_bf16<2><<<dim3(HID / BN, TOK / BM), 256, 0, stream>>>(
        interm, FFN, sdw, nullptr, FFN, out, HID, counts + NEXP, nullptr, nullptr);

    // routed experts: top-2 gather / weighted scatter-add (stream-serialized)
    for (int e = 0; e < NEXP; ++e) {
        gemm_bf16<1><<<dim3(FFN / BN, TOK / BM), 256, 0, stream>>>(
            hid, HID, gate_w + (size_t)e * FFN * HID, up_w + (size_t)e * FFN * HID,
            HID, interm, FFN, counts + e, tok + (size_t)e * TOK, nullptr);
        gemm_bf16<2><<<dim3(HID / BN, TOK / BM), 256, 0, stream>>>(
            interm, FFN, down_w + (size_t)e * HID * FFN, nullptr,
            FFN, out, HID, counts + e, tok + (size_t)e * TOK, wtbuf + (size_t)e * TOK);
    }
}
